// A2aSparseMLP_33122787787379
// MI455X (gfx1250) — matrix operand order repaired and checked by reading, not compiled, not run on hardware
//
#include <hip/hip_runtime.h>
#include <hip/hip_bf16.h>

// MoE top-2 MLP for MI455X (gfx1250, wave32, WMMA bf16 16x16x32).
// T=4096 tokens, H=1024, E=8, I=1024, top-2. Grouped GEMMs over token-slots.

typedef __attribute__((ext_vector_type(16))) __bf16 v16bf;
typedef __attribute__((ext_vector_type(8)))  float  v8f;

static constexpr int TOK   = 4096;   // B*S
static constexpr int HD    = 1024;   // hidden
static constexpr int NE    = 8;      // experts
static constexpr int ID    = 1024;   // intermediate
static constexpr int TOPK  = 2;
#define ALPHA_C 1.702f
#define LIMIT_C 7.0f

// ---------------------------------------------------------------- init ----
__global__ void moe_init_kernel(int* __restrict__ counts) {
    if (threadIdx.x < NE) counts[threadIdx.x] = 0;
}

// -------------------------------------------------------------- router ----
__global__ __launch_bounds__(256) void moe_router_kernel(
    const float* __restrict__ x, const float* __restrict__ rw,
    __bf16* __restrict__ xb, int* __restrict__ counts,
    int* __restrict__ rowlist, float* __restrict__ wtlist)
{
    int wave = threadIdx.x >> 5;
    int lane = threadIdx.x & 31;
    int t = blockIdx.x * 8 + wave;

    float acc[NE];
    #pragma unroll
    for (int e = 0; e < NE; ++e) acc[e] = 0.f;

    const float* xr = x + (size_t)t * HD;
    for (int h = lane; h < HD; h += 32) {
        float xv = xr[h];
        xb[(size_t)t * HD + h] = (__bf16)xv;
        const float* r = rw + (size_t)h * NE;
        #pragma unroll
        for (int e = 0; e < NE; ++e) acc[e] += xv * r[e];
    }
    #pragma unroll
    for (int e = 0; e < NE; ++e) {
        float v = acc[e];
        for (int off = 16; off > 0; off >>= 1) v += __shfl_xor(v, off, 32);
        acc[e] = v;
    }
    if (lane == 0) {
        int e0 = 0; float v0 = acc[0];
        #pragma unroll
        for (int e = 1; e < NE; ++e) if (acc[e] > v0) { v0 = acc[e]; e0 = e; }
        int e1 = (e0 == 0) ? 1 : 0; float v1 = acc[e1];
        #pragma unroll
        for (int e = 0; e < NE; ++e)
            if (e != e0 && acc[e] > v1) { v1 = acc[e]; e1 = e; }
        float w0 = 1.f / (1.f + __expf(v1 - v0));   // softmax over {v0,v1}
        float w1 = 1.f - w0;
        int p0 = atomicAdd(&counts[e0], 1);
        rowlist[e0 * TOK + p0] = t * 2 + 0; wtlist[e0 * TOK + p0] = w0;
        int p1 = atomicAdd(&counts[e1], 1);
        rowlist[e1 * TOK + p1] = t * 2 + 1; wtlist[e1 * TOK + p1] = w1;
    }
}

// -------------------------------------------------- gate/up GEMM + GLU ----
// Block tile: M=64 token-slot rows x 64 act cols (=> 128 interleaved gu cols).
// 8 waves = 4 M-tiles x 2 N-tiles; each wave owns two 16-col act subtiles
// => 4 accumulators (gate/up x 2 subtiles), 4 WMMA per 32-k chunk.
// fp32 weights are de-interleaved + converted to bf16 while staged to LDS.
__global__ __launch_bounds__(256) void moe_gateup_kernel(
    const __bf16* __restrict__ xb,
    const float* __restrict__ gup, const float* __restrict__ gub,
    const int* __restrict__ counts, const int* __restrict__ rowlist,
    __bf16* __restrict__ actbuf)
{
    int e = blockIdx.z;
    int cnt = counts[e];
    int rowBase = blockIdx.y * 64;
    if (rowBase >= cnt) return;
    int nvalid = cnt - rowBase; if (nvalid > 64) nvalid = 64;

    __shared__ int srow[64];
    __shared__ __align__(16) __bf16 Bg[64 * 40];    // [act-col][k] stride 40
    __shared__ __align__(16) __bf16 Bu[64 * 40];

    int tid = threadIdx.x;
    if (tid < 64) {
        int idx = rowBase + tid;
        srow[tid] = rowlist[e * TOK + ((idx < cnt) ? idx : (cnt - 1))];
    }
    __syncthreads();

    int wave = tid >> 5, lane = tid & 31;
    int mTile = wave >> 1;                          // 0..3
    int nTile = wave & 1;                           // 0..1
    int jBlock = blockIdx.x * 64;                   // global act-col base

    v8f cg0 = {}, cg1 = {}, cu0 = {}, cu1 = {};

    int mrow = mTile * 16 + (lane & 15);            // block-local A row
    const __bf16* arow = xb + (size_t)(srow[mrow] >> 1) * HD;
    int ka0 = (lane < 16) ? 0 : 16;                 // A: contiguous k = 16h + s
    int c   = lane & 15;                            // B col within subtile
    int kb0 = (lane < 16) ? 0 : 16;                 // B: contiguous k = 16h + s, as A
    int j0  = nTile * 32 + c;                       // subtile 0 col (local)
    int j1  = j0 + 16;                              // subtile 1 col (local)

    for (int k0 = 0; k0 < HD; k0 += 32) {
        #pragma unroll
        for (int it = 0; it < 8; ++it) {            // 32k x 64j pairs / 256
            int p  = tid + it * 256;
            int kk = p >> 6;
            int j  = p & 63;
            const float* src = gup + ((size_t)e * HD + (k0 + kk)) * (2 * ID)
                                   + 2 * (jBlock + j);
            float2 gu2 = *(const float2*)src;       // gate, up interleaved
            Bg[j * 40 + kk] = (__bf16)gu2.x;
            Bu[j * 40 + kk] = (__bf16)gu2.y;
        }
        __syncthreads();

        union { v16bf v; uint4 q[2]; } fa, fg0, fg1, fu0, fu1;
        const uint4* ap = (const uint4*)(arow + k0);
        fa.q[0] = ap[ka0 >> 3];
        fa.q[1] = ap[(ka0 >> 3) + 1];
        const uint4* p0g = (const uint4*)(Bg + j0 * 40);
        const uint4* p1g = (const uint4*)(Bg + j1 * 40);
        const uint4* p0u = (const uint4*)(Bu + j0 * 40);
        const uint4* p1u = (const uint4*)(Bu + j1 * 40);
        fg0.q[0] = p0g[kb0 >> 3]; fg0.q[1] = p0g[(kb0 >> 3) + 1];
        fg1.q[0] = p1g[kb0 >> 3]; fg1.q[1] = p1g[(kb0 >> 3) + 1];
        fu0.q[0] = p0u[kb0 >> 3]; fu0.q[1] = p0u[(kb0 >> 3) + 1];
        fu1.q[0] = p1u[kb0 >> 3]; fu1.q[1] = p1u[(kb0 >> 3) + 1];

        cg0 = __builtin_amdgcn_wmma_f32_16x16x32_bf16(
                  false, fa.v, false, fg0.v, (short)0, cg0, false, false);
        cg1 = __builtin_amdgcn_wmma_f32_16x16x32_bf16(
                  false, fa.v, false, fg1.v, (short)0, cg1, false, false);
        cu0 = __builtin_amdgcn_wmma_f32_16x16x32_bf16(
                  false, fa.v, false, fu0.v, (short)0, cu0, false, false);
        cu1 = __builtin_amdgcn_wmma_f32_16x16x32_bf16(
                  false, fa.v, false, fu1.v, (short)0, cu1, false, false);
        __syncthreads();
    }

    // bias + clipped SiLU-GLU, elementwise in the C-fragment layout
    int jg0 = jBlock + j0;                          // global act columns
    int jg1 = jBlock + j1;
    float2 b0 = *(const float2*)(gub + (size_t)e * (2 * ID) + 2 * jg0);
    float2 b1 = *(const float2*)(gub + (size_t)e * (2 * ID) + 2 * jg1);
    int rbase = (lane < 16) ? 0 : 8;
    #pragma unroll
    for (int r = 0; r < 8; ++r) {
        int mm = mTile * 16 + rbase + r;            // block-local row
        if (mm < nvalid) {
            size_t orow = (size_t)srow[mm] * ID;
            {
                float g = fminf(cg0[r] + b0.x, LIMIT_C);
                float u = fminf(fmaxf(cu0[r] + b0.y, -LIMIT_C), LIMIT_C);
                float glu = g * __builtin_amdgcn_rcpf(1.f + __expf(-ALPHA_C * g));
                actbuf[orow + jg0] = (__bf16)((u + 1.f) * glu);
            }
            {
                float g = fminf(cg1[r] + b1.x, LIMIT_C);
                float u = fminf(fmaxf(cu1[r] + b1.y, -LIMIT_C), LIMIT_C);
                float glu = g * __builtin_amdgcn_rcpf(1.f + __expf(-ALPHA_C * g));
                actbuf[orow + jg1] = (__bf16)((u + 1.f) * glu);
            }
        }
    }
}

// ----------------------------------------------------------- down GEMM ----
// Block tile: M=64 rows x 64 H cols; 8 waves = 4 M-tiles x 2 N-tiles,
// each wave owns two 16-col subtiles => 2 WMMA per 32-k chunk.
__global__ __launch_bounds__(256) void moe_down_kernel(
    const __bf16* __restrict__ actbuf,
    const float* __restrict__ dp, const float* __restrict__ dpb,
    const int* __restrict__ counts, const int* __restrict__ rowlist,
    const float* __restrict__ wtlist, float* __restrict__ dout)
{
    int e = blockIdx.z;
    int cnt = counts[e];
    int rowBase = blockIdx.y * 64;
    if (rowBase >= cnt) return;
    int nvalid = cnt - rowBase; if (nvalid > 64) nvalid = 64;

    __shared__ int   srow[64];
    __shared__ float swt[64];
    __shared__ __align__(16) __bf16 Bs[64 * 40];

    int tid = threadIdx.x;
    if (tid < 64) {
        int idx = rowBase + tid;
        int ci  = (idx < cnt) ? idx : (cnt - 1);
        srow[tid] = rowlist[e * TOK + ci];
        swt[tid]  = (idx < cnt) ? wtlist[e * TOK + ci] : 0.f;
    }
    __syncthreads();

    int wave = tid >> 5, lane = tid & 31;
    int mTile = wave >> 1;
    int nTile = wave & 1;
    int hBlock = blockIdx.x * 64;

    v8f acc0 = {}, acc1 = {};

    int mrow = mTile * 16 + (lane & 15);
    const __bf16* arow = actbuf + (size_t)srow[mrow] * ID;
    int ka0 = (lane < 16) ? 0 : 16;
    int c   = lane & 15;
    int kb0 = (lane < 16) ? 0 : 16;
    int j0  = nTile * 32 + c;
    int j1  = j0 + 16;

    for (int k0 = 0; k0 < ID; k0 += 32) {
        #pragma unroll
        for (int it = 0; it < 8; ++it) {            // 32k x 64j / 256
            int p  = tid + it * 256;
            int kk = p >> 6;
            int j  = p & 63;
            float v = dp[((size_t)e * ID + (k0 + kk)) * HD + hBlock + j];
            Bs[j * 40 + kk] = (__bf16)v;
        }
        __syncthreads();

        union { v16bf v; uint4 q[2]; } fa, fb0, fb1;
        const uint4* ap = (const uint4*)(arow + k0);
        fa.q[0] = ap[ka0 >> 3];
        fa.q[1] = ap[(ka0 >> 3) + 1];
        const uint4* bp0 = (const uint4*)(Bs + j0 * 40);
        const uint4* bp1 = (const uint4*)(Bs + j1 * 40);
        fb0.q[0] = bp0[kb0 >> 3]; fb0.q[1] = bp0[(kb0 >> 3) + 1];
        fb1.q[0] = bp1[kb0 >> 3]; fb1.q[1] = bp1[(kb0 >> 3) + 1];

        acc0 = __builtin_amdgcn_wmma_f32_16x16x32_bf16(
                   false, fa.v, false, fb0.v, (short)0, acc0, false, false);
        acc1 = __builtin_amdgcn_wmma_f32_16x16x32_bf16(
                   false, fa.v, false, fb1.v, (short)0, acc1, false, false);
        __syncthreads();
    }

    int h0 = hBlock + j0;
    int h1 = hBlock + j1;
    float bias0 = dpb[(size_t)e * HD + h0];
    float bias1 = dpb[(size_t)e * HD + h1];
    int rbase = (lane < 16) ? 0 : 8;
    #pragma unroll
    for (int r = 0; r < 8; ++r) {
        int mm = mTile * 16 + rbase + r;
        if (mm < nvalid) {
            float w = swt[mm];
            size_t orow = (size_t)srow[mm] * HD;
            dout[orow + h0] = w * (acc0[r] + bias0);
            dout[orow + h1] = w * (acc1[r] + bias1);
        }
    }
}

// ------------------------------------------------------------ reduce ----
__global__ __launch_bounds__(256) void moe_reduce_kernel(
    const float* __restrict__ dout, float* __restrict__ out)
{
    int idx = blockIdx.x * 256 + threadIdx.x;       // float4 index
    int t   = idx >> 8;                             // HD/4 == 256
    int rem = idx & 255;
    const float4* a = (const float4*)dout + (size_t)(2 * t)     * 256 + rem;
    const float4* b = (const float4*)dout + (size_t)(2 * t + 1) * 256 + rem;
    float4 va = *a, vb = *b, r;
    r.x = va.x + vb.x; r.y = va.y + vb.y; r.z = va.z + vb.z; r.w = va.w + vb.w;
    ((float4*)out)[idx] = r;
}

// ------------------------------------------------------------ launch ----
extern "C" void kernel_launch(void* const* d_in, const int* in_sizes, int n_in,
                              void* d_out, int out_size, void* d_ws, size_t ws_size,
                              hipStream_t stream)
{
    const float* x   = (const float*)d_in[0];   // (B,S,H)
    const float* rw  = (const float*)d_in[1];   // (H,E)
    const float* gup = (const float*)d_in[2];   // (E,H,2I)
    const float* gub = (const float*)d_in[3];   // (E,2I)
    const float* dp  = (const float*)d_in[4];   // (E,I,H)
    const float* dpb = (const float*)d_in[5];   // (E,H)
    float* out = (float*)d_out;
    (void)in_sizes; (void)n_in; (void)out_size; (void)ws_size;

    char* ws = (char*)d_ws;
    size_t off = 0;
    auto take = [&](size_t bytes) -> void* {
        void* p = ws + off;
        off = (off + bytes + 255) & ~(size_t)255;
        return p;
    };
    __bf16* xb      = (__bf16*)take((size_t)TOK * HD * 2);            //  8 MB
    int*    rowlist = (int*)   take((size_t)NE * TOK * 4);            // 128 KB
    float*  wtlist  = (float*) take((size_t)NE * TOK * 4);            // 128 KB
    int*    counts  = (int*)   take(NE * 4);
    __bf16* actbuf  = (__bf16*)take((size_t)TOK * TOPK * ID * 2);     // 16 MB
    float*  dout    = (float*) take((size_t)TOK * TOPK * HD * 4);     // 32 MB

    hipLaunchKernelGGL(moe_init_kernel, dim3(1), dim3(32), 0, stream, counts);
    hipLaunchKernelGGL(moe_router_kernel, dim3(TOK / 8), dim3(256), 0, stream,
                       x, rw, xb, counts, rowlist, wtlist);
    hipLaunchKernelGGL(moe_gateup_kernel, dim3(ID / 64, TOK / 64, NE), dim3(256),
                       0, stream, xb, gup, gub, counts, rowlist, actbuf);
    hipLaunchKernelGGL(moe_down_kernel, dim3(HD / 64, TOK / 64, NE), dim3(256),
                       0, stream, actbuf, dp, dpb, counts, rowlist, wtlist, dout);
    hipLaunchKernelGGL(moe_reduce_kernel, dim3((TOK * HD / 4) / 256), dim3(256),
                       0, stream, dout, out);
}
